// RSCA_Resnet_26233660244751
// MI455X (gfx1250) — matrix operand order repaired and checked by reading, not compiled, not run on hardware
//
#include <hip/hip_runtime.h>
#include <math.h>

typedef __bf16 bf16_t;
typedef __attribute__((ext_vector_type(16))) __bf16 v16bf;
typedef __attribute__((ext_vector_type(8)))  float  v8f;
typedef __attribute__((ext_vector_type(4)))  unsigned int u32x4;

union BV {
  v16bf v;
  u32x4 q[2];
};

// ---------------------------------------------------------------------------
// Implicit-GEMM conv using v_wmma_f32_16x16x32_bf16, 16(co) x 64(px) per wave.
//   act : NHWC bf16  (2, Hin, Win, Cin), Cin % 32 == 0
//   wgt : bf16 [CoPad][KH][KW][Cin]   (rows >= CoutActual zero-padded)
//   D = W(16xK) x Patches(Kx16) ; epilogue: scale/shift (+res) (+relu)
// A-frag (16x32 bf16): lane L -> M=L&15, K = (L<16?0:16)+[0,16) contiguous
// B-frag (32x16 bf16): lane L -> N=L&15, K = (L<16?0:16)+[0,16) contiguous
// C/D   : lane L -> N=L&15 ; VGPR j -> M = j + (L<16?0:8)
// One A-fragment feeds 4 back-to-back WMMAs (4 N-tiles) -> 4x weight reuse.
// ---------------------------------------------------------------------------
template <bool RELU, bool RES, bool OUTF32>
__global__ __launch_bounds__(32) void conv_wmma(
    const bf16_t* __restrict__ act, const bf16_t* __restrict__ wgt,
    const float* __restrict__ scale, const float* __restrict__ shift,
    const bf16_t* __restrict__ res, void* __restrict__ outp,
    int Hin, int Win, int Cin, int Hout, int Wout,
    int CoutStore, int CoutActual, int KH, int KW, int stride, int pad)
{
  const int lane = threadIdx.x;
  const int lo = lane & 15;
  const int hi = lane >> 4;
  const int co0 = blockIdx.x * 16;
  const int img = blockIdx.z;
  const int total = Hout * Wout;
  const int Ktot = KH * KW * Cin;

  // 4 spatial sub-tiles of 16 positions; tail-masked for total % 64 != 0.
  int ohh[4], oww[4];
  bool pv[4];
#pragma unroll
  for (int t = 0; t < 4; ++t) {
    int p = blockIdx.y * 64 + t * 16 + lo;
    pv[t] = p < total;
    if (!pv[t]) p = total - 1;
    ohh[t] = p / Wout;
    oww[t] = p % Wout;
  }

  v8f acc[4];
#pragma unroll
  for (int t = 0; t < 4; ++t) acc[t] = (v8f){0.f,0.f,0.f,0.f,0.f,0.f,0.f,0.f};

  const bf16_t* arow = wgt + (size_t)(co0 + lo) * Ktot + (hi ? 16 : 0);
  const int bko = hi * 16;

  for (int kh = 0; kh < KH; ++kh) {
    for (int kw = 0; kw < KW; ++kw) {
      const bf16_t* ap = arow + (kh * KW + kw) * Cin;
      __builtin_prefetch(ap + Cin, 0, 0);   // next filter tap (global_prefetch_b8)
      const bf16_t* bp[4];
      bool ok[4];
#pragma unroll
      for (int t = 0; t < 4; ++t) {
        const int ih = ohh[t] * stride + kh - pad;
        const int iw = oww[t] * stride + kw - pad;
        ok[t] = pv[t] && ((unsigned)ih < (unsigned)Hin) && ((unsigned)iw < (unsigned)Win);
        bp[t] = act + (((size_t)img * Hin + ih) * Win + iw) * Cin + bko;
      }
      for (int cc = 0; cc < Cin; cc += 32) {
        BV a;
        a.q[0] = *(const u32x4*)(ap + cc);
        a.q[1] = *(const u32x4*)(ap + cc + 8);
#pragma unroll
        for (int t = 0; t < 4; ++t) {
          BV b;
          if (ok[t]) {
            b.q[0] = *(const u32x4*)(bp[t] + cc);
            b.q[1] = *(const u32x4*)(bp[t] + cc + 8);
          } else {
            u32x4 z = {0u, 0u, 0u, 0u};
            b.q[0] = z;
            b.q[1] = z;
          }
          acc[t] = __builtin_amdgcn_wmma_f32_16x16x32_bf16(
              false, a.v, false, b.v, (short)0, acc[t], false, false);
        }
      }
    }
  }

  const int cob = co0 + hi * 8;
#pragma unroll
  for (int t = 0; t < 4; ++t) {
    if (!pv[t]) continue;
    const size_t obase =
        (((size_t)img * Hout + ohh[t]) * Wout + oww[t]) * (size_t)CoutStore + cob;
#pragma unroll
    for (int j = 0; j < 8; ++j) {
      const int co = cob + j;
      if (co >= CoutActual) continue;
      float v = acc[t][j] * scale[co] + shift[co];
      if (RES) v += (float)res[obase + j];
      if (RELU) v = v > 0.f ? v : 0.f;
      if (OUTF32) ((float*)outp)[obase + j] = v;
      else        ((bf16_t*)outp)[obase + j] = (bf16_t)v;
    }
  }
}

// ---------------------------------------------------------------------------
// Stem: direct 7x7/s2 conv, Cin=3, NCHW f32 in -> NHWC bf16 out (bias+BN+ReLU)
// ---------------------------------------------------------------------------
__global__ void stem_conv_kernel(const float* __restrict__ x,
                                 const float* __restrict__ w,
                                 const float* __restrict__ scale,
                                 const float* __restrict__ shift,
                                 bf16_t* __restrict__ out)
{
  const int co = threadIdx.x;   // 64
  const int ow = blockIdx.x;    // 192
  const int oh = blockIdx.y;    // 192
  const int b  = blockIdx.z;    // 2
  float acc = 0.f;
  for (int ci = 0; ci < 3; ++ci)
    for (int kh = 0; kh < 7; ++kh) {
      const int ih = oh * 2 + kh - 3;
      if ((unsigned)ih >= 384u) continue;
      for (int kw = 0; kw < 7; ++kw) {
        const int iw = ow * 2 + kw - 3;
        if ((unsigned)iw >= 384u) continue;
        acc += x[(((size_t)b * 3 + ci) * 384 + ih) * 384 + iw] *
               w[((co * 3 + ci) * 7 + kh) * 7 + kw];
      }
    }
  float v = acc * scale[co] + shift[co];
  v = v > 0.f ? v : 0.f;
  out[(((size_t)b * 192 + oh) * 192 + ow) * 64 + co] = (bf16_t)v;
}

// maxpool 3x3 s2 p1 : (2,192,192,64) -> (2,96,96,64) NHWC bf16
__global__ void maxpool_kernel(const bf16_t* __restrict__ in, bf16_t* __restrict__ out)
{
  const int c = threadIdx.x;
  const int ow = blockIdx.x, oh = blockIdx.y, b = blockIdx.z;
  float m = -3.4e38f;
  for (int di = 0; di < 3; ++di) {
    const int ih = oh * 2 + di - 1;
    if ((unsigned)ih >= 192u) continue;
    for (int dj = 0; dj < 3; ++dj) {
      const int iw = ow * 2 + dj - 1;
      if ((unsigned)iw >= 192u) continue;
      m = fmaxf(m, (float)in[(((size_t)b * 192 + ih) * 192 + iw) * 64 + c]);
    }
  }
  out[(((size_t)b * 96 + oh) * 96 + ow) * 64 + c] = (bf16_t)m;
}

// LCAU mix: per-pixel softmax over 25 predicted weights, 5x5 adaptive smoothing
__global__ void lcau_mix_kernel(const float* __restrict__ kc,
                                const bf16_t* __restrict__ x,
                                bf16_t* __restrict__ y, int H, int W, int C)
{
  const int w0 = blockIdx.x, h0 = blockIdx.y, b = blockIdx.z;
  const float* kp = kc + (((size_t)b * H + h0) * W + w0) * 25;
  float s[25];
  float mx = -3.4e38f;
  for (int l = 0; l < 25; ++l) { s[l] = kp[l]; mx = fmaxf(mx, s[l]); }
  float sum = 0.f;
  for (int l = 0; l < 25; ++l) { s[l] = __expf(s[l] - mx); sum += s[l]; }
  const float inv = 1.f / sum;
  for (int c = threadIdx.x; c < C; c += blockDim.x) {
    float acc = 0.f;
    for (int i = 0; i < 5; ++i) {
      const int ih = h0 + i - 2;
      if ((unsigned)ih >= (unsigned)H) continue;
      for (int j = 0; j < 5; ++j) {
        const int iw = w0 + j - 2;
        if ((unsigned)iw >= (unsigned)W) continue;
        acc += s[i * 5 + j] * (float)x[(((size_t)b * H + ih) * W + iw) * C + c];
      }
    }
    y[(((size_t)b * H + h0) * W + w0) * C + c] = (bf16_t)(acc * inv);
  }
}

// nearest x2 upsample + skip add (NHWC bf16)
__global__ void up2_add_kernel(const bf16_t* __restrict__ m,
                               const bf16_t* __restrict__ skip,
                               bf16_t* __restrict__ out,
                               int H2, int W2, int C, long total)
{
  long i = (long)blockIdx.x * blockDim.x + threadIdx.x;
  if (i >= total) return;
  const int c = (int)(i % C); long t = i / C;
  const int ow = (int)(t % W2); t /= W2;
  const int oh = (int)(t % H2);
  const int b = (int)(t / H2);
  const float v =
      (float)m[(((size_t)b * (H2 >> 1) + (oh >> 1)) * (W2 >> 1) + (ow >> 1)) * C + c] +
      (float)skip[i];
  out[i] = (bf16_t)v;
}

// final: m NHWC f32 (2,96,96,32) -> sigmoid(up2) -> NCHW f32 (2,32,192,192)
__global__ void final_kernel(const float* __restrict__ m, float* __restrict__ out, long total)
{
  long i = (long)blockIdx.x * blockDim.x + threadIdx.x;
  if (i >= total) return;
  const int ow = (int)(i % 192); long t = i / 192;
  const int oh = (int)(t % 192); t /= 192;
  const int co = (int)(t % 32);
  const int b = (int)(t / 32);
  const float v = m[(((size_t)b * 96 + (oh >> 1)) * 96 + (ow >> 1)) * 32 + co];
  out[i] = 1.f / (1.f + __expf(-v));
}

// OIHW f32 -> bf16 [CoPad][kh][kw][ci]
__global__ void prep_weight_kernel(const float* __restrict__ w, bf16_t* __restrict__ o,
                                   int Co, int Ci, int KH, int KW, int CoPad)
{
  long i = (long)blockIdx.x * blockDim.x + threadIdx.x;
  const long total = (long)CoPad * KH * KW * Ci;
  if (i >= total) return;
  const int ci = (int)(i % Ci); long t = i / Ci;
  const int kw = (int)(t % KW); t /= KW;
  const int kh = (int)(t % KH);
  const int co = (int)(t / KH);
  float v = 0.f;
  if (co < Co) v = w[(((size_t)co * Ci + ci) * KH + kh) * KW + kw];
  o[i] = (bf16_t)v;
}

// LCAU w2 (Co,4C,1,1) f32 -> W2eff bf16 [Co][C], W2eff[co][c] = sum_r w2[co][4c+r]
__global__ void prep_w2eff_kernel(const float* __restrict__ w2, bf16_t* __restrict__ o,
                                  int Co, int C)
{
  long i = (long)blockIdx.x * blockDim.x + threadIdx.x;
  if (i >= (long)Co * C) return;
  const int c = (int)(i % C);
  const int co = (int)(i / C);
  float s = 0.f;
  for (int r = 0; r < 4; ++r) s += w2[(size_t)co * 4 * C + c * 4 + r];
  o[i] = (bf16_t)s;
}

// fold (optional bias) + (optional BN) into per-channel scale/shift
__global__ void prep_ss_kernel(const float* __restrict__ g, const float* __restrict__ bb,
                               const float* __restrict__ mm, const float* __restrict__ vv,
                               const float* __restrict__ bias,
                               float* __restrict__ scale, float* __restrict__ shift, int C)
{
  const int c = blockIdx.x * blockDim.x + threadIdx.x;
  if (c >= C) return;
  float sc = 1.f, sh = 0.f;
  if (g) {
    const float inv = g[c] * rsqrtf(vv[c] + 1e-5f);
    sc = inv;
    sh = bb[c] - mm[c] * inv + (bias ? bias[c] * inv : 0.f);
  } else if (bias) {
    sh = bias[c];
  }
  scale[c] = sc;
  shift[c] = sh;
}

// ---------------------------------------------------------------------------
extern "C" void kernel_launch(void* const* d_in, const int* in_sizes, int n_in,
                              void* d_out, int out_size, void* d_ws, size_t ws_size,
                              hipStream_t stream)
{
  (void)in_sizes; (void)n_in; (void)out_size; (void)ws_size;

  char* base = (char*)d_ws;
  size_t off = 0;
  auto alloc = [&](size_t bytes) -> void* {
    off = (off + 255) & ~(size_t)255;
    void* p = base + off;
    off += bytes;
    return p;
  };
  auto F = [&](int i) -> const float* { return (const float*)d_in[i]; };

  // -------- d_in index map (jax tree order: sorted dict keys, list order) ----
  // params.layers:
  //  L0B0: bn1 0..3(b,g,m,v) bn2 4..7 w1 8 w2 9 ; L0B1: 10..13,14..17,w1 18,w2 19
  //  L1B0: bn1 20..23 bn2 24..27 dbn 28..31 down_b 32 down_w 33 w1 34 w2 35
  //  L1B1: 36..39,40..43,w1 44,w2 45
  //  L2B0: 46..49,50..53,54..57,58,59,60,61 ; L2B1: 62..65,66..69,70,71
  //  L3B0: 72..75,76..79,80..83,84,85,86,87 ; L3B1: 88..91,92..95,96,97
  // params.lcau[i]: b1 98+4i, b2 99+4i, w1 100+4i, w2 101+4i
  // stem_b 114 ; stem_bn b 115 g 116 m 117 v 118 ; stem_w 119 ; x 120

  struct Blk { int bn1, bn2, dbn, downb, downw, w1, w2, cin, cout, stride; };
  const Blk blks[8] = {
    { 0,  4, -1, -1, -1,  8,  9,  64,  64, 1},
    {10, 14, -1, -1, -1, 18, 19,  64,  64, 1},
    {20, 24, 28, 32, 33, 34, 35,  64, 128, 2},
    {36, 40, -1, -1, -1, 44, 45, 128, 128, 1},
    {46, 50, 54, 58, 59, 60, 61, 128, 256, 2},
    {62, 66, -1, -1, -1, 70, 71, 256, 256, 1},
    {72, 76, 80, 84, 85, 86, 87, 256, 512, 2},
    {88, 92, -1, -1, -1, 96, 97, 512, 512, 1},
  };

  auto prep_w = [&](int idx, int Co, int Ci, int KH, int KW, int CoPad) -> bf16_t* {
    bf16_t* o = (bf16_t*)alloc((size_t)CoPad * KH * KW * Ci * sizeof(bf16_t));
    const long total = (long)CoPad * KH * KW * Ci;
    prep_weight_kernel<<<dim3((unsigned)((total + 255) / 256)), 256, 0, stream>>>(
        F(idx), o, Co, Ci, KH, KW, CoPad);
    return o;
  };
  auto prep_ss = [&](const float* g, const float* b, const float* m, const float* v,
                     const float* bias, int C, float** sc, float** sh) {
    *sc = (float*)alloc(C * sizeof(float));
    *sh = (float*)alloc(C * sizeof(float));
    prep_ss_kernel<<<dim3((C + 63) / 64), 64, 0, stream>>>(g, b, m, v, bias, *sc, *sh, C);
  };
  auto launch_conv = [&](const bf16_t* in, const bf16_t* wq, const float* sc,
                         const float* sh, const bf16_t* res, void* out, int Hin, int Cin,
                         int Hout, int CoStore, int CoAct, int CoPad, int KH, int KW,
                         int s, int pad, int mode /*0 plain,1 relu,2 relu+res,3 f32out*/) {
    dim3 grid(CoPad / 16, (Hout * Hout + 63) / 64, 2), block(32);
    switch (mode) {
      case 0:
        conv_wmma<false, false, false><<<grid, block, 0, stream>>>(
            in, wq, sc, sh, res, out, Hin, Hin, Cin, Hout, Hout, CoStore, CoAct, KH, KW, s, pad);
        break;
      case 1:
        conv_wmma<true, false, false><<<grid, block, 0, stream>>>(
            in, wq, sc, sh, res, out, Hin, Hin, Cin, Hout, Hout, CoStore, CoAct, KH, KW, s, pad);
        break;
      case 2:
        conv_wmma<true, true, false><<<grid, block, 0, stream>>>(
            in, wq, sc, sh, res, out, Hin, Hin, Cin, Hout, Hout, CoStore, CoAct, KH, KW, s, pad);
        break;
      default:
        conv_wmma<false, false, true><<<grid, block, 0, stream>>>(
            in, wq, sc, sh, res, out, Hin, Hin, Cin, Hout, Hout, CoStore, CoAct, KH, KW, s, pad);
        break;
    }
  };

  // ---------------- stem + maxpool ----------------
  float *stem_sc, *stem_sh;
  prep_ss(F(116), F(115), F(117), F(118), F(114), 64, &stem_sc, &stem_sh);
  bf16_t* stem_out = (bf16_t*)alloc((size_t)2 * 192 * 192 * 64 * sizeof(bf16_t));
  stem_conv_kernel<<<dim3(192, 192, 2), 64, 0, stream>>>(F(120), F(119), stem_sc,
                                                         stem_sh, stem_out);
  bf16_t* pool = (bf16_t*)alloc((size_t)2 * 96 * 96 * 64 * sizeof(bf16_t));
  maxpool_kernel<<<dim3(96, 96, 2), 64, 0, stream>>>(stem_out, pool);

  // ---------------- residual blocks ----------------
  const bf16_t* feats[4];
  const bf16_t* cur = pool;
  int H = 96;
  for (int bi = 0; bi < 8; ++bi) {
    const Blk& B = blks[bi];
    const int Ho = H / B.stride;

    bf16_t* w1q = prep_w(B.w1, B.cout, B.cin, 3, 3, B.cout);
    float *sc1, *sh1;
    prep_ss(F(B.bn1 + 1), F(B.bn1 + 0), F(B.bn1 + 2), F(B.bn1 + 3), nullptr, B.cout, &sc1, &sh1);
    bf16_t* w2q = prep_w(B.w2, B.cout, B.cout, 3, 3, B.cout);
    float *sc2, *sh2;
    prep_ss(F(B.bn2 + 1), F(B.bn2 + 0), F(B.bn2 + 2), F(B.bn2 + 3), nullptr, B.cout, &sc2, &sh2);

    bf16_t* t = (bf16_t*)alloc((size_t)2 * Ho * Ho * B.cout * sizeof(bf16_t));
    launch_conv(cur, w1q, sc1, sh1, nullptr, t, H, B.cin, Ho,
                B.cout, B.cout, B.cout, 3, 3, B.stride, 1, 1);

    const bf16_t* idn = cur;
    if (B.dbn >= 0) {
      bf16_t* wdq = prep_w(B.downw, B.cout, B.cin, 1, 1, B.cout);
      float *scd, *shd;
      prep_ss(F(B.dbn + 1), F(B.dbn + 0), F(B.dbn + 2), F(B.dbn + 3), F(B.downb),
              B.cout, &scd, &shd);
      bf16_t* dn = (bf16_t*)alloc((size_t)2 * Ho * Ho * B.cout * sizeof(bf16_t));
      launch_conv(cur, wdq, scd, shd, nullptr, dn, H, B.cin, Ho,
                  B.cout, B.cout, B.cout, 1, 1, B.stride, 0, 0);
      idn = dn;
    }

    bf16_t* o = (bf16_t*)alloc((size_t)2 * Ho * Ho * B.cout * sizeof(bf16_t));
    launch_conv(t, w2q, sc2, sh2, idn, o, Ho, B.cout, Ho,
                B.cout, B.cout, B.cout, 3, 3, 1, 1, 2);
    cur = o;
    H = Ho;
    if (bi & 1) feats[bi >> 1] = o;
  }

  // ---------------- LCAU decoder ----------------
  const int lci[4][4] = {{98, 99, 100, 101}, {102, 103, 104, 105},
                         {106, 107, 108, 109}, {110, 111, 112, 113}};
  const int lcout[4] = {256, 128, 64, 32};
  const bf16_t* dcur = feats[3];
  int Hc = 12, Cc = 512;
  for (int i = 0; i < 4; ++i) {
    const int b1 = lci[i][0], b2 = lci[i][1], w1 = lci[i][2], w2 = lci[i][3];
    const int Co = lcout[i];

    // kernel prediction: conv3x3 -> 25 channels (CoPad 32), f32 out
    bf16_t* w1q = prep_w(w1, 25, Cc, 3, 3, 32);
    float *sck, *shk;
    prep_ss(nullptr, nullptr, nullptr, nullptr, F(b1), 25, &sck, &shk);
    float* kc = (float*)alloc((size_t)2 * Hc * Hc * 25 * sizeof(float));
    launch_conv(dcur, w1q, sck, shk, nullptr, kc, Hc, Cc, Hc, 25, 25, 32, 3, 3, 1, 1, 3);

    // softmax + 5x5 content-adaptive smoothing (r-invariant -> done once)
    bf16_t* y = (bf16_t*)alloc((size_t)2 * Hc * Hc * Cc * sizeof(bf16_t));
    lcau_mix_kernel<<<dim3(Hc, Hc, 2), 128, 0, stream>>>(kc, dcur, y, Hc, Hc, Cc);

    // 1x1 conv with W2eff = sum over r of w2
    bf16_t* w2q = (bf16_t*)alloc((size_t)Co * Cc * sizeof(bf16_t));
    {
      const long tt = (long)Co * Cc;
      prep_w2eff_kernel<<<dim3((unsigned)((tt + 255) / 256)), 256, 0, stream>>>(
          F(w2), w2q, Co, Cc);
    }
    float *sc2, *sh2;
    prep_ss(nullptr, nullptr, nullptr, nullptr, F(b2), Co, &sc2, &sh2);

    if (i < 3) {
      bf16_t* m = (bf16_t*)alloc((size_t)2 * Hc * Hc * Co * sizeof(bf16_t));
      launch_conv(y, w2q, sc2, sh2, nullptr, m, Hc, Cc, Hc, Co, Co, Co, 1, 1, 1, 0, 0);
      const int H2 = Hc * 2;
      bf16_t* d = (bf16_t*)alloc((size_t)2 * H2 * H2 * Co * sizeof(bf16_t));
      const long tot = (long)2 * H2 * H2 * Co;
      up2_add_kernel<<<dim3((unsigned)((tot + 255) / 256)), 256, 0, stream>>>(
          m, feats[2 - i], d, H2, H2, Co, tot);
      dcur = d;
      Hc = H2;
      Cc = Co;
    } else {
      float* m = (float*)alloc((size_t)2 * 96 * 96 * 32 * sizeof(float));
      launch_conv(y, w2q, sc2, sh2, nullptr, m, 96, 64, 96, 32, 32, 32, 1, 1, 1, 0, 3);
      const long tot = (long)2 * 32 * 192 * 192;
      final_kernel<<<dim3((unsigned)((tot + 255) / 256)), 256, 0, stream>>>(
          m, (float*)d_out, tot);
    }
  }
}
